// Scaser_27754078667308
// MI455X (gfx1250) — matrix operand order repaired and checked by reading, not compiled, not run on hardware
//
#include <hip/hip_runtime.h>
#include <hip/hip_bf16.h>
#include <cstdint>

// ---- model constants (fixed by the reference) ----
#define TT    4
#define LSEQ  50
#define DIM   256
#define BB    512
#define NH    8
#define NV    4
#define FEAT  1424          // NV*DIM + NH*LSEQ = 1024 + 400

typedef __attribute__((ext_vector_type(16))) __bf16 v16bf;
typedef __attribute__((ext_vector_type(8)))  float  v8f;

union ABfrag { uint4 q[2]; v16bf v; };

// Closed-form 4-step LIF (tau=2, v_th=1, hard reset, constant input c):
// returns mean-over-T spike value. Matches reference numerics exactly
// (x*0.5f == x/2.0 in binary FP).
__device__ __forceinline__ float lif4_mean(float c) {
  float v = 0.f, cnt = 0.f;
#pragma unroll
  for (int t = 0; t < TT; ++t) {
    v = v + (c - v) * 0.5f;               // charge
    float s = (v >= 1.0f) ? 1.0f : 0.0f;  // fire (atan_spike fwd = heaviside)
    cnt += s;
    v = v * (1.0f - s);                   // hard reset (detached)
  }
  return cnt * 0.25f;
}

// ---- 1) embedding gather + cast to bf16:  Xb[b*L+l][d] ----
__global__ void __launch_bounds__(256)
gather_cast(const int* __restrict__ seq, const float* __restrict__ emb,
            __bf16* __restrict__ Xb) {
  int idx = blockIdx.x;            // b*LSEQ + l
  int d   = threadIdx.x;
  int item = seq[idx];
  Xb[(size_t)idx * DIM + d] = (__bf16)emb[(size_t)item * DIM + d];
}

// ---- 2) vertical conv in f32 (tiny, precision-preserving): feat[b][v*256+d] ----
__global__ void __launch_bounds__(256)
convv_kernel(const int* __restrict__ seq, const float* __restrict__ emb,
             const float* __restrict__ wv, float* __restrict__ feat) {
  int b = blockIdx.x, d = threadIdx.x;
  __shared__ int s[LSEQ];
  if (d < LSEQ) s[d] = seq[b * LSEQ + d];
  __syncthreads();
  float a0 = 0.f, a1 = 0.f, a2 = 0.f, a3 = 0.f;
  for (int l = 0; l < LSEQ; ++l) {
    float x = emb[(size_t)s[l] * DIM + d];
    a0 = fmaf(wv[l],            x, a0);
    a1 = fmaf(wv[LSEQ + l],     x, a1);
    a2 = fmaf(wv[2 * LSEQ + l], x, a2);
    a3 = fmaf(wv[3 * LSEQ + l], x, a3);
  }
  float* fr = feat + (size_t)b * FEAT;
  fr[d] = a0; fr[DIM + d] = a1; fr[2 * DIM + d] = a2; fr[3 * DIM + d] = a3;
}

// ---- 3) per-k weight pack: 8 real rows + 8 zero rows, bf16, row-major [16 x Kk] ----
__global__ void pack_w(const float* __restrict__ src, __bf16* __restrict__ dst,
                       int Kk) {
  int n = blockIdx.x * blockDim.x + threadIdx.x;
  if (n >= 16 * Kk) return;
  int m = n / Kk;                  // padded row; src layout [h][i][d] == same flat index
  dst[n] = (m < NH) ? (__bf16)src[n] : (__bf16)0.f;
}

// ---- 4) conv_h as WMMA GEMM + fused LIF + width-mean ----
// One wave owns one batch row b for kernel-size k:
//   C[16 x 16] tiles over columns j (output positions), K = (k+1)*256.
// Key addressing fact: the global K index is kc*32, and the im2col column
// X[b, j..j+k, :] is CONTIGUOUS in memory, so both the A and B streams are
// plain 64-byte-stride pointer walks — no per-iteration index math.
// A fragment (16x32 bf16, ISA layout): lane<16 row m=lane holds K {0..7,16..23},
// lane>=16 row m=lane-16 holds K {8..15,24..31}.
// B fragment (32x16 bf16): lane%16 = column n; lane<16 holds K {0..7,16..23},
// lane>=16 holds K {8..15,24..31} (same order as A).
// C fragment: lanes 0..15, VGPR r  <->  (M=r, N=lane) — our 8 valid filter rows.
__global__ void __launch_bounds__(256)
convh_wmma(const __bf16* __restrict__ Xb, const __bf16* __restrict__ Wp,
           float* __restrict__ feat, int k, int width, int Kk) {
  int lane = threadIdx.x & 31;
  int b    = blockIdx.x * 8 + (threadIdx.x >> 5);
  if (b >= BB) return;
  const bool hi  = lane >= 16;
  const int  nlo = lane & 15;
  const int  tiles = (width + 15) >> 4;

  float hsum[8];
#pragma unroll
  for (int r = 0; r < 8; ++r) hsum[r] = 0.f;

  // Per-lane fixed base pointers (A restarts every tile; B depends on column j).
  const __bf16* abase = Wp + (size_t)nlo * Kk + (hi ? 8 : 0);
  const __bf16* xrow  = Xb + (size_t)b * (LSEQ * DIM) + (hi ? 8 : 0);
  const int nkc = Kk >> 5;               // (k+1)*8 chunks of 32 K-values

  for (int jt = 0; jt < tiles; ++jt) {
    int j0 = jt << 4;
    int j  = j0 + nlo;                   // this lane's output column
    if (j > width - 1) j = width - 1;    // clamp: padded columns load in-range, discarded later
    v8f acc = {};
    const __bf16* ap = abase;
    const __bf16* bp = xrow + (size_t)j * DIM;
    for (int kc = 0; kc < nkc; ++kc) {
      ABfrag A, B;
      A.q[0] = *(const uint4*)(ap);        // halves 0..7
      A.q[1] = *(const uint4*)(ap + 16);   // halves 8..15 (K +16)
      B.q[0] = *(const uint4*)(bp);        // halves 0..7
      B.q[1] = *(const uint4*)(bp + 16);   // halves 8..15 (K +16)
      if ((kc & 7) == 0) __builtin_prefetch(bp + DIM, 0, 1);  // next X row ahead
      acc = __builtin_amdgcn_wmma_f32_16x16x32_bf16(
          /*neg_a=*/false, A.v, /*neg_b=*/false, B.v,
          /*c_mod=*/(short)0, acc, /*reuse_a=*/false, /*reuse_b=*/false);
      ap += 32;                            // 64-byte K stride, pure pointer walk
      bp += 32;
    }
#pragma unroll
    for (int r = 0; r < 8; ++r) {
      bool valid = (!hi) && (j0 + nlo < width);   // real filter row & real column
      hsum[r] += valid ? lif4_mean(acc[r]) : 0.f;
    }
  }

  float invw = 1.0f / (float)width;
#pragma unroll
  for (int r = 0; r < 8; ++r) {
    float f = hsum[r];
#pragma unroll
    for (int off = 8; off >= 1; off >>= 1) f += __shfl_xor(f, off, 16);
    if (lane == 0)
      feat[(size_t)b * FEAT + NV * DIM + k * NH + r] = f * invw;  // deterministic, no atomics
  }
}

// ---- 5) final FC in f32: out[b][d] = fc_b[d] + feat[b][:] . fc_w[d][:] ----
__global__ void __launch_bounds__(256)
fc_kernel(const float* __restrict__ feat, const float* __restrict__ fcw,
          const float* __restrict__ fcb, float* __restrict__ out) {
  int b = blockIdx.x, d = threadIdx.x;
  __shared__ float row[FEAT];
  for (int i = d; i < FEAT; i += DIM) row[i] = feat[(size_t)b * FEAT + i];
  __syncthreads();
  const float* wrow = fcw + (size_t)d * FEAT;
  float s = fcb[d];
  for (int i = 0; i < FEAT; ++i) s = fmaf(row[i], wrow[i], s);
  out[(size_t)b * DIM + d] = s;
}

extern "C" void kernel_launch(void* const* d_in, const int* in_sizes, int n_in,
                              void* d_out, int out_size, void* d_ws, size_t ws_size,
                              hipStream_t stream) {
  (void)in_sizes; (void)out_size; (void)ws_size;

  const int*   item_seq = (const int*)d_in[0];
  const float* emb      = (const float*)d_in[1];
  const float* wv       = (const float*)d_in[2];
  const float* convh[LSEQ];
  const float* fcw;
  const float* fcb;

  if (n_in >= 3 + LSEQ + 2) {
    // conv_h_ws tuple flattened into 50 separate inputs
    for (int k = 0; k < LSEQ; ++k) convh[k] = (const float*)d_in[3 + k];
    fcw = (const float*)d_in[3 + LSEQ];
    fcb = (const float*)d_in[4 + LSEQ];
  } else {
    // conv_h_ws packed into one concatenated buffer
    const float* base = (const float*)d_in[3];
    size_t off = 0;
    for (int k = 0; k < LSEQ; ++k) {
      convh[k] = base + off;
      off += (size_t)NH * (k + 1) * DIM;
    }
    fcw = (const float*)d_in[4];
    fcb = (const float*)d_in[5];
  }

  // workspace layout (~25.3 MB total)
  char*   ws    = (char*)d_ws;
  __bf16* Xb    = (__bf16*)ws;                                       // 13,107,200 B
  size_t  xbB   = (size_t)BB * LSEQ * DIM * sizeof(__bf16);
  __bf16* Wpack = (__bf16*)(ws + xbB);                               // 10,444,800 B
  size_t  wpB   = (size_t)16 * DIM * (LSEQ * (LSEQ + 1) / 2) * sizeof(__bf16);
  float*  feat  = (float*)(ws + xbB + wpB);                          //  2,916,352 B

  gather_cast<<<BB * LSEQ, DIM, 0, stream>>>(item_seq, emb, Xb);
  convv_kernel<<<BB, DIM, 0, stream>>>(item_seq, emb, wv, feat);

  size_t offW = 0;
  for (int k = 0; k < LSEQ; ++k) {
    int Kk = (k + 1) * DIM;
    int nelem = 16 * Kk;
    pack_w<<<(nelem + 255) / 256, 256, 0, stream>>>(convh[k], Wpack + offW, Kk);
    offW += (size_t)16 * Kk;
  }

  offW = 0;
  for (int k = 0; k < LSEQ; ++k) {
    int Kk = (k + 1) * DIM;
    int width = LSEQ - k;
    convh_wmma<<<BB / 8, 256, 0, stream>>>(Xb, Wpack + offW, feat, k, width, Kk);
    offW += (size_t)16 * Kk;
  }

  fc_kernel<<<BB, DIM, 0, stream>>>(feat, fcw, fcb, (float*)d_out);
}
